// SeqClassifier_50044958933282
// MI455X (gfx1250) — matrix operand order repaired and checked by reading, not compiled, not run on hardware
//
#include <hip/hip_runtime.h>

// ---------------------------------------------------------------------------
// 3x LSTM (B=256, T=1024, D=4, H=100) -> concat -> FC(300->12) -> ReLU
// CDNA5 (gfx1250) wave32 WMMA implementation.
//
// Per LSTM, per timestep:  gates[256,512p] = state[256,128p] @ Wc^T
//   state row = [ h (100 f16) | x_t (4) | 1.0 (bias slot) | 0 pad ]  (K = 128)
//   Wc row n  = [ W_hh[n][:]  | W_ih[n][:] | b_ih[n]+b_hh[n] | 0 ]  (N = 4*128)
// Bias rides in the GEMM via the constant-1 K slot, so WMMA accumulators
// start from inline zero (no splat movs). Each wave owns a 16-row batch
// strip; batch rows are independent across the recurrence, so the T-loop
// needs no inter-wave barriers.
// ---------------------------------------------------------------------------

typedef __attribute__((ext_vector_type(16))) _Float16 v16h;
typedef __attribute__((ext_vector_type(8)))  _Float16 v8h;
typedef __attribute__((ext_vector_type(4)))  _Float16 v4h;
typedef __attribute__((ext_vector_type(8)))  float    v8f;
typedef __attribute__((ext_vector_type(4)))  float    v4f;

#define T_STEPS 1024
#define D_IN    4
#define HID     100
#define BSLOT   (HID + D_IN)   // K index 104 carries the constant 1.0 (bias)
#define GSTR    128            // padded per-gate column stride (N = 4*128 = 512)
#define KP      128            // padded K (100 h + 4 x + 1 bias + 23 zero)
#define LROW    136            // LDS row stride in halves (272B -> bank-conflict-free)
#define WPB     4              // waves per block
#define BPL     4              // blocks per LSTM
#define RPB     (WPB * 16)     // batch rows per block (64)

// gfx1250 has native V_TANH_F32 (TRANS op); fall back to exp-based if the
// builtin isn't exposed under either name.
__device__ __forceinline__ float tanh_native(float x) {
#if __has_builtin(__builtin_amdgcn_tanhf)
    return __builtin_amdgcn_tanhf(x);
#elif __has_builtin(__builtin_amdgcn_tanh_f32)
    return __builtin_amdgcn_tanh_f32(x);
#else
    return 2.0f * __builtin_amdgcn_rcpf(1.0f + __expf(-2.0f * x)) - 1.0f;
#endif
}
__device__ __forceinline__ float sigmoid_native(float x) {
    // sigmoid(x) = 0.5*tanh(x/2) + 0.5  -> 1 trans + 1 mul + 1 fma
    return fmaf(0.5f, tanh_native(0.5f * x), 0.5f);
}

__launch_bounds__(128, 1)
__global__ void lstm_scan_kernel(
    const float* __restrict__ x1, const float* __restrict__ x2, const float* __restrict__ x3,
    const float* __restrict__ Wih1, const float* __restrict__ Whh1,
    const float* __restrict__ bih1, const float* __restrict__ bhh1,
    const float* __restrict__ Wih2, const float* __restrict__ Whh2,
    const float* __restrict__ bih2, const float* __restrict__ bhh2,
    const float* __restrict__ Wih3, const float* __restrict__ Whh3,
    const float* __restrict__ bih3, const float* __restrict__ bhh3,
    float* __restrict__ ws)   // ws: [3][256][100] f32 last hidden states
{
    __shared__ __align__(16) _Float16 sWc[512 * LROW];   // combined weights+bias (f16)
    __shared__ __align__(16) _Float16 sH [RPB * LROW];   // state rows [h | x | 1 | pad]

    const int lstm = blockIdx.x / BPL;
    const int blk  = blockIdx.x % BPL;

    const float* X   = (lstm == 0) ? x1   : (lstm == 1) ? x2   : x3;
    const float* Wih = (lstm == 0) ? Wih1 : (lstm == 1) ? Wih2 : Wih3;
    const float* Whh = (lstm == 0) ? Whh1 : (lstm == 1) ? Whh2 : Whh3;
    const float* bi  = (lstm == 0) ? bih1 : (lstm == 1) ? bih2 : bih3;
    const float* bh  = (lstm == 0) ? bhh1 : (lstm == 1) ? bhh2 : bhh3;

    const int tid = threadIdx.x;

    // ---- stage combined weights (f32 -> f16) into LDS --------------------
    for (int idx = tid; idx < 512 * KP; idx += 128) {
        int n = idx / KP, k = idx - n * KP;
        int gt = n >> 7, hc = n & 127;
        float v = 0.0f;
        if (hc < HID) {
            int no = gt * HID + hc;                 // PyTorch gate order i,f,g,o
            if (k < HID)              v = Whh[no * HID + k];
            else if (k < BSLOT)       v = Wih[no * D_IN + (k - HID)];
            else if (k == BSLOT)      v = bi[no] + bh[no];   // bias via 1.0 K-slot
        }
        sWc[n * LROW + k] = (_Float16)v;
    }
    // zero state; plant the constant-1 bias slot
    for (int idx = tid; idx < RPB * KP; idx += 128) {
        int r = idx / KP, k = idx - r * KP;
        sH[r * LROW + k] = (k == BSLOT) ? (_Float16)1.0f : (_Float16)0.0f;
    }
    __syncthreads();

    // ---- per-wave geometry ----------------------------------------------
    const int wave = tid >> 5;
    const int lane = tid & 31;
    const int col  = lane & 15;       // D/B layout: column within tile
    const int kh   = lane >> 4;       // half-wave selector
    const int rowA = wave * 16 + col; // A layout: row = lane%16
    const int rowD = wave * 16 + 8 * kh;
    const int bRow0 = blk * RPB;      // global batch row base of this block

    float c_reg[7][8];                // cell state, WMMA-D layout, nh tiles 0..6
    #pragma unroll
    for (int i = 0; i < 7; ++i)
        #pragma unroll
        for (int r = 0; r < 8; ++r) c_reg[i][r] = 0.0f;

    const _Float16* aRow = &sH[rowA * LROW];
    const int myRow = wave * 16 + lane;            // for x injection (lanes 0..15)
    const size_t xBase = ((size_t)(bRow0 + myRow)) * T_STEPS * D_IN;

    // software pipeline: preload x_0
    v4f xv = {0.f, 0.f, 0.f, 0.f};
    if (lane < 16) xv = *(const v4f*)&X[xBase];

    // ---- sequential scan over T -----------------------------------------
    for (int t = 0; t < T_STEPS; ++t) {
        // inject x_t into the state's K-slot [100..103] for this wave's rows
        if (lane < 16) {
            v4h hv;
            hv[0] = (_Float16)xv[0]; hv[1] = (_Float16)xv[1];
            hv[2] = (_Float16)xv[2]; hv[3] = (_Float16)xv[3];
            *(v4h*)&sH[myRow * LROW + HID] = hv;
            // prefetch x_{t+1}: global latency hides under the WMMA body
            if (t + 1 < T_STEPS) xv = *(const v4f*)&X[xBase + (size_t)(t + 1) * D_IN];
        }

        // A fragments: 4 K-slabs of 32, reused across all 28 output tiles
        v16h a[4];
        #pragma unroll
        for (int ks = 0; ks < 4; ++ks) {
            v8h lo = *(const v8h*)&aRow[ks * 32 + kh * 8];
            v8h hi = *(const v8h*)&aRow[ks * 32 + 16 + kh * 8];
            a[ks] = __builtin_shufflevector(lo, hi,
                     0, 1, 2, 3, 4, 5, 6, 7, 8, 9, 10, 11, 12, 13, 14, 15);
        }

        #pragma unroll
        for (int nh = 0; nh < 7; ++nh) {
            v8f acc[4];
            #pragma unroll
            for (int g = 0; g < 4; ++g)
                #pragma unroll
                for (int r = 0; r < 8; ++r) acc[g][r] = 0.0f;   // inline-0 SRC2

            #pragma unroll
            for (int ks = 0; ks < 4; ++ks) {
                #pragma unroll
                for (int g = 0; g < 4; ++g) {
                    const _Float16* wp =
                        &sWc[(g * GSTR + nh * 16 + col) * LROW + ks * 32 + kh * 8];
                    v8h lo = *(const v8h*)wp;
                    v8h hi = *(const v8h*)(wp + 16);
                    v16h bf = __builtin_shufflevector(lo, hi,
                               0, 1, 2, 3, 4, 5, 6, 7, 8, 9, 10, 11, 12, 13, 14, 15);
                    acc[g] = __builtin_amdgcn_wmma_f32_16x16x32_f16(
                                 false, a[ks], false, bf, (short)0, acc[g],
                                 false, false);
                }
            }
            // gate nonlinearity + state update (i,f,g,o)
            int hc = nh * 16 + col;
            #pragma unroll
            for (int r = 0; r < 8; ++r) {
                float si = sigmoid_native(acc[0][r]);
                float sf = sigmoid_native(acc[1][r]);
                float tg = tanh_native(acc[2][r]);
                float so = sigmoid_native(acc[3][r]);
                float c  = sf * c_reg[nh][r] + si * tg;
                c_reg[nh][r] = c;
                float hnew = so * tanh_native(c);
                if (hc < HID)
                    sH[(rowD + r) * LROW + hc] = (_Float16)hnew;
            }
        }
    }
    __syncthreads();

    // ---- export last hidden state (f32) to workspace --------------------
    for (int idx = tid; idx < RPB * HID; idx += 128) {
        int r = idx / HID, hc = idx - r * HID;
        ws[((size_t)lstm * 256 + bRow0 + r) * HID + hc] = (float)sH[r * LROW + hc];
    }
}

// ---------------------------------------------------------------------------
// FC: out[b][j] = relu( concat(h1,h2,h3)[b] . fc_W[j] + fc_b[j] )
// ---------------------------------------------------------------------------
__global__ void fc_relu_kernel(const float* __restrict__ ws,
                               const float* __restrict__ fcW,
                               const float* __restrict__ fcb,
                               float* __restrict__ out)
{
    int j = blockIdx.x;      // 0..11
    int b = threadIdx.x;     // 0..255
    float acc = fcb[j];
    #pragma unroll
    for (int l = 0; l < 3; ++l) {
        const float* hv = &ws[((size_t)l * 256 + b) * HID];
        const float* wv = &fcW[j * (3 * HID) + l * HID];
        #pragma unroll 4
        for (int k = 0; k < HID; ++k) acc += hv[k] * wv[k];
    }
    out[b * 12 + j] = fmaxf(acc, 0.0f);
}

extern "C" void kernel_launch(void* const* d_in, const int* in_sizes, int n_in,
                              void* d_out, int out_size, void* d_ws, size_t ws_size,
                              hipStream_t stream) {
    const float* x1   = (const float*)d_in[0];
    const float* x2   = (const float*)d_in[1];
    const float* x3   = (const float*)d_in[2];
    const float* Wih1 = (const float*)d_in[3];
    const float* Whh1 = (const float*)d_in[4];
    const float* bih1 = (const float*)d_in[5];
    const float* bhh1 = (const float*)d_in[6];
    const float* Wih2 = (const float*)d_in[7];
    const float* Whh2 = (const float*)d_in[8];
    const float* bih2 = (const float*)d_in[9];
    const float* bhh2 = (const float*)d_in[10];
    const float* Wih3 = (const float*)d_in[11];
    const float* Whh3 = (const float*)d_in[12];
    const float* bih3 = (const float*)d_in[13];
    const float* bhh3 = (const float*)d_in[14];
    const float* fcW  = (const float*)d_in[15];
    const float* fcb  = (const float*)d_in[16];
    float* ws = (float*)d_ws;

    lstm_scan_kernel<<<dim3(3 * BPL), dim3(128), 0, stream>>>(
        x1, x2, x3,
        Wih1, Whh1, bih1, bhh1,
        Wih2, Whh2, bih2, bhh2,
        Wih3, Whh3, bih3, bhh3,
        ws);

    fc_relu_kernel<<<dim3(12), dim3(256), 0, stream>>>(ws, fcW, fcb, (float*)d_out);
}
